// MultiheadSparseAttn_9929964388669
// MI455X (gfx1250) — matrix operand order repaired and checked by reading, not compiled, not run on hardware
//
#include <hip/hip_runtime.h>
#include <hip/hip_bf16.h>
#include <math.h>

#define SEQ     8192
#define BATCH   2
#define EMBED   512
#define HEADS   8
#define DH      64
#define BLOCKSZ 64
#define NBLK    128          // SEQ / BLOCKSZ
#define NRAND   3
#define MGATH   8            // 2 global + 3 window + 3 random
#define KEYS    512          // MGATH * BLOCKSZ

typedef __bf16 bf16_t;
typedef __attribute__((ext_vector_type(16))) __bf16 v16bf;
typedef __attribute__((ext_vector_type(8)))  __bf16 v8bf;
typedef __attribute__((ext_vector_type(8)))  float  v8f;

// ---------------------------------------------------------------------------
// Fragment loaders per CDNA5 ISA 7.12.2 (wave32):
//  A 16x32 bf16: lane L -> row m=L%16; halves 0..7 = K k0+8*(L>=16)+0..7,
//                halves 8..15 = same +16. Two contiguous 16B loads.
//  B 32x16 bf16 from B^T [N][K] row-major: lane L -> row n=L%16; halves 0..7 =
//                K k0+8*(L>=16)+0..7, halves 8..15 = same +16. Two 16B loads.
// ---------------------------------------------------------------------------
__device__ __forceinline__ v16bf load_a_frag(const bf16_t* A, int lda, int kbase) {
  const int lane = threadIdx.x & 31;
  const int m  = lane & 15;
  const int k0 = kbase + ((lane >> 4) << 3);
  const bf16_t* p = A + m * lda + k0;
  v8bf lo = *(const v8bf*)(p);
  v8bf hi = *(const v8bf*)(p + 16);
  v16bf r;
#pragma unroll
  for (int i = 0; i < 8; ++i) { r[i] = lo[i]; r[i + 8] = hi[i]; }
  return r;
}

__device__ __forceinline__ v16bf load_b_frag(const bf16_t* BT, int ldb, int kbase) {
  const int lane = threadIdx.x & 31;
  const int n  = lane & 15;
  const int k0 = kbase + ((lane >> 4) << 3);
  const bf16_t* p = BT + n * ldb + k0;
  v8bf lo = *(const v8bf*)(p);
  v8bf hi = *(const v8bf*)(p + 16);
  v16bf r;
#pragma unroll
  for (int i = 0; i < 8; ++i) { r[i] = lo[i]; r[i + 8] = hi[i]; }
  return r;
}

__device__ __forceinline__ v8f wmma_bf(v16bf a, v16bf b, v8f c) {
  // (neg_a, A, neg_b, B, c_mod, C, reuse_a, reuse_b)
  return __builtin_amdgcn_wmma_f32_16x16x32_bf16(false, a, false, b, (short)0, c,
                                                 false, false);
}

// ---------------------------------------------------------------------------
// Stage 0a: x (S,B,E) f32 -> xbf [t][b][s][e] bf16 for t in {q,k,v}
// ---------------------------------------------------------------------------
__global__ void cvt_x_kernel(const float* __restrict__ q,
                             const float* __restrict__ k,
                             const float* __restrict__ v,
                             bf16_t* __restrict__ xbf) {
  const long per = (long)BATCH * SEQ * EMBED;
  long idx = (long)blockIdx.x * 256 + threadIdx.x;
  if (idx >= 3 * per) return;
  int t = (int)(idx / per);
  long r = idx - (long)t * per;
  int b = (int)(r / ((long)SEQ * EMBED));
  long r2 = r - (long)b * SEQ * EMBED;
  int s = (int)(r2 / EMBED);
  int e = (int)(r2 - (long)s * EMBED);
  const float* src = (t == 0) ? q : (t == 1) ? k : v;
  xbf[idx] = (bf16_t)src[((long)s * BATCH + b) * EMBED + e];
}

// ---------------------------------------------------------------------------
// Stage 0b: weights -> bf16, pre-packed as B^T [n][k]
//   wbf[t][n=h*64+d][e] = w_t[h][e][d]      (t = 0..2)
//   wobf[n][k]          = out_w[n][k]       (reference uses out_w.T already)
// ---------------------------------------------------------------------------
__global__ void cvt_w_kernel(const float* __restrict__ qw,
                             const float* __restrict__ kw,
                             const float* __restrict__ vw,
                             const float* __restrict__ ow,
                             bf16_t* __restrict__ wbf,
                             bf16_t* __restrict__ wobf) {
  int idx = blockIdx.x * 256 + threadIdx.x;  // 4 * 512 * 512
  int t = idx >> 18;
  int r = idx & ((1 << 18) - 1);
  if (t < 3) {
    int n = r >> 9;
    int e = r & 511;
    int h = n >> 6;
    int d = n & 63;
    const float* w = (t == 0) ? qw : (t == 1) ? kw : vw;
    wbf[idx] = (bf16_t)w[((long)h * EMBED + e) * DH + d];
  } else {
    wobf[r] = (bf16_t)ow[r];
  }
}

// ---------------------------------------------------------------------------
// Stage 1: QKV projection GEMM.  grid = (128 seq-tiles, 8 heads, b*3+t)
// block = 128 (4 waves), each wave a 32x32 sub-tile of the 64x64 output tile.
//   Q -> Qbuf [b][h][nb][q][d]   (row-major per block; A for scores)
//   K -> Kbuf [b][h][nb][key][d] (row-major per block; B^T for scores)
//   V -> Vtbuf[b][h][nb][d][key] (transposed per block; B^T for PV)
// ---------------------------------------------------------------------------
__global__ void __launch_bounds__(128)
proj_kernel(const bf16_t* __restrict__ xbf, const bf16_t* __restrict__ wbf,
            const float* __restrict__ qb, const float* __restrict__ kb,
            const float* __restrict__ vb,
            bf16_t* __restrict__ Qbuf, bf16_t* __restrict__ Kbuf,
            bf16_t* __restrict__ Vtbuf) {
  const int nb = blockIdx.x;
  const int h  = blockIdx.y;
  const int bz = blockIdx.z;
  const int b = bz / 3, t = bz % 3;
  const int wave = threadIdx.x >> 5;
  const int lane = threadIdx.x & 31;
  const int wm = wave >> 1, wn = wave & 1;

  const bf16_t* A  = xbf + (((long)t * BATCH + b) * SEQ + (long)nb * BLOCKSZ) * EMBED;
  const bf16_t* BT = wbf + ((long)t * EMBED + h * DH) * EMBED;

  v8f acc[2][2] = {};
  for (int ks = 0; ks < EMBED / 32; ++ks) {
    const int k0 = ks * 32;
    v16bf a0 = load_a_frag(A + (wm * 32) * EMBED, EMBED, k0);
    v16bf a1 = load_a_frag(A + (wm * 32 + 16) * EMBED, EMBED, k0);
    v16bf b0 = load_b_frag(BT + (wn * 32) * EMBED, EMBED, k0);
    v16bf b1 = load_b_frag(BT + (wn * 32 + 16) * EMBED, EMBED, k0);
    acc[0][0] = wmma_bf(a0, b0, acc[0][0]);
    acc[0][1] = wmma_bf(a0, b1, acc[0][1]);
    acc[1][0] = wmma_bf(a1, b0, acc[1][0]);
    acc[1][1] = wmma_bf(a1, b1, acc[1][1]);
  }

  const float* bias = (t == 0) ? qb : (t == 1) ? kb : vb;  // [H][DH]
  const long blkbase = (long)(b * HEADS + h) * NBLK + nb;
#pragma unroll
  for (int i = 0; i < 2; ++i) {
#pragma unroll
    for (int j = 0; j < 2; ++j) {
      const int nloc = wn * 32 + j * 16 + (lane & 15);        // d
      const float bv = bias[h * DH + nloc];
#pragma unroll
      for (int r = 0; r < 8; ++r) {
        const int mloc = wm * 32 + i * 16 + r + ((lane >> 4) << 3);  // row in 64
        const float val = acc[i][j][r] + bv;
        if (t == 0)
          Qbuf[(blkbase * BLOCKSZ + mloc) * DH + nloc] = (bf16_t)val;
        else if (t == 1)
          Kbuf[(blkbase * BLOCKSZ + mloc) * DH + nloc] = (bf16_t)val;
        else
          Vtbuf[(blkbase * DH + nloc) * BLOCKSZ + mloc] = (bf16_t)val;
      }
    }
  }
}

// ---------------------------------------------------------------------------
// Stage 2: block-sparse attention. grid = (NB, H, B), block = 256 (8 waves).
// Wave w owns gathered key-block w: computes the 64x64 score tile (16 WMMA
// accumulators), softmax stats via shfl + LDS, writes unnormalized bf16 probs
// to LDS, then all waves run the 64x512 x 512x64 PV GEMM; 1/rowsum is folded
// into the output store.
// ---------------------------------------------------------------------------
__global__ void __launch_bounds__(256)
attn_kernel(const bf16_t* __restrict__ Qbuf, const bf16_t* __restrict__ Kbuf,
            const bf16_t* __restrict__ Vtbuf, const int* __restrict__ rand_idx,
            bf16_t* __restrict__ AObuf) {
  const int nb = blockIdx.x, h = blockIdx.y, b = blockIdx.z;
  const int tid = threadIdx.x;
  const int wave = tid >> 5, lane = tid & 31;

  __shared__ bf16_t Qs[BLOCKSZ][DH + 8];       // +8 halves keeps 16B row align
  __shared__ bf16_t Ps[BLOCKSZ][KEYS + 8];
  __shared__ float  redmax[BLOCKSZ][MGATH];
  __shared__ float  redsum[BLOCKSZ][MGATH];
  __shared__ int    sblk[MGATH];

  if (tid < MGATH) {
    int v;
    if (tid < 2)      v = tid;                               // global blocks
    else if (tid < 5) v = (nb + tid - 3 + NBLK) % NBLK;      // window -1,0,+1
    else              v = rand_idx[nb * NRAND + (tid - 5)];  // random
    sblk[tid] = v;
  }

  const long bh = (long)(b * HEADS + h);
  const bf16_t* Qg = Qbuf + (bh * NBLK + nb) * (long)(BLOCKSZ * DH);
  for (int i = tid; i < BLOCKSZ * DH; i += 256)
    Qs[i >> 6][i & 63] = Qg[i];
  __syncthreads();

  // ---- scores: S_w = Qb (64x64) @ K_blk^T (64x64) ----
  const bf16_t* Kb = Kbuf + (bh * NBLK + sblk[wave]) * (long)(BLOCKSZ * DH);
  v8f s[4][4] = {};
#pragma unroll
  for (int kd = 0; kd < DH; kd += 32) {
    v16bf af[4], bg[4];
#pragma unroll
    for (int mt = 0; mt < 4; ++mt) af[mt] = load_a_frag(&Qs[mt * 16][0], DH + 8, kd);
#pragma unroll
    for (int nt = 0; nt < 4; ++nt) bg[nt] = load_b_frag(Kb + (nt * 16) * DH, DH, kd);
#pragma unroll
    for (int mt = 0; mt < 4; ++mt)
#pragma unroll
      for (int nt = 0; nt < 4; ++nt)
        s[mt][nt] = wmma_bf(af[mt], bg[nt], s[mt][nt]);
  }

  const float scale = 0.125f;  // 1/sqrt(64)

  // ---- per-wave row max over this key block ----
#pragma unroll
  for (int mt = 0; mt < 4; ++mt) {
#pragma unroll
    for (int r = 0; r < 8; ++r) {
      float mx = -3.0e38f;
#pragma unroll
      for (int nt = 0; nt < 4; ++nt) mx = fmaxf(mx, s[mt][nt][r]);
      mx *= scale;
#pragma unroll
      for (int off = 1; off < 16; off <<= 1)
        mx = fmaxf(mx, __shfl_xor(mx, off, 32));
      const int m = mt * 16 + r + ((lane >> 4) << 3);
      if ((lane & 15) == 0) redmax[m][wave] = mx;
    }
  }
  __syncthreads();

  // ---- global max, exp, partial sums, bf16 probs to LDS ----
#pragma unroll
  for (int mt = 0; mt < 4; ++mt) {
#pragma unroll
    for (int r = 0; r < 8; ++r) {
      const int m = mt * 16 + r + ((lane >> 4) << 3);
      float gm = redmax[m][0];
#pragma unroll
      for (int j = 1; j < MGATH; ++j) gm = fmaxf(gm, redmax[m][j]);
      float partial = 0.f;
#pragma unroll
      for (int nt = 0; nt < 4; ++nt) {
        const float p = __expf(s[mt][nt][r] * scale - gm);
        partial += p;
        Ps[m][wave * BLOCKSZ + nt * 16 + (lane & 15)] = (bf16_t)p;
      }
#pragma unroll
      for (int off = 1; off < 16; off <<= 1)
        partial += __shfl_xor(partial, off, 32);
      if ((lane & 15) == 0) redsum[m][wave] = partial;
    }
  }
  __syncthreads();

  // ---- PV: out (64x64) = P (64x512) @ Vg (512x64); wave -> 2 C tiles ----
  const int mt  = wave >> 1;
  const int nt0 = (wave & 1) * 2;
  v8f o0 = {}, o1 = {};
  for (int ks = 0; ks < KEYS / 32; ++ks) {
    const int blk  = ks >> 1;
    const int koff = (ks & 1) * 32;
    v16bf a = load_a_frag(&Ps[mt * 16][0], KEYS + 8, ks * 32);
    const bf16_t* Vb = Vtbuf + (bh * NBLK + sblk[blk]) * (long)(DH * BLOCKSZ);
    v16bf b0 = load_b_frag(Vb + (nt0 * 16) * BLOCKSZ, BLOCKSZ, koff);
    v16bf b1 = load_b_frag(Vb + ((nt0 + 1) * 16) * BLOCKSZ, BLOCKSZ, koff);
    o0 = wmma_bf(a, b0, o0);
    o1 = wmma_bf(a, b1, o1);
  }

  // ---- normalize by row sum, store flat [b*S+s][h*64+d] bf16 ----
  const long rowbase = (long)b * SEQ + (long)nb * BLOCKSZ;
#pragma unroll
  for (int r = 0; r < 8; ++r) {
    const int m = mt * 16 + r + ((lane >> 4) << 3);
    float sum = redsum[m][0];
#pragma unroll
    for (int j = 1; j < MGATH; ++j) sum += redsum[m][j];
    const float inv = 1.f / sum;
    const int d0 = nt0 * 16 + (lane & 15);
    AObuf[(rowbase + m) * EMBED + h * DH + d0]      = (bf16_t)(o0[r] * inv);
    AObuf[(rowbase + m) * EMBED + h * DH + d0 + 16] = (bf16_t)(o1[r] * inv);
  }
}

// ---------------------------------------------------------------------------
// Stage 3: output projection. grid = (256 row-tiles, 8 col-tiles), 4 waves.
// C = AObuf (16384x512 bf16) @ out_w^T + out_b, scattered to (S,B,E) f32.
// ---------------------------------------------------------------------------
__global__ void __launch_bounds__(128)
outproj_kernel(const bf16_t* __restrict__ AObuf, const bf16_t* __restrict__ wobf,
               const float* __restrict__ outb, float* __restrict__ out) {
  const int mtile = blockIdx.x;
  const int ntile = blockIdx.y;
  const int wave = threadIdx.x >> 5, lane = threadIdx.x & 31;
  const int wm = wave >> 1, wn = wave & 1;

  const bf16_t* A  = AObuf + (long)(mtile * BLOCKSZ) * EMBED;
  const bf16_t* BT = wobf + (long)(ntile * 64) * EMBED;

  v8f acc[2][2] = {};
  for (int ks = 0; ks < EMBED / 32; ++ks) {
    const int k0 = ks * 32;
    v16bf a0 = load_a_frag(A + (wm * 32) * EMBED, EMBED, k0);
    v16bf a1 = load_a_frag(A + (wm * 32 + 16) * EMBED, EMBED, k0);
    v16bf b0 = load_b_frag(BT + (wn * 32) * EMBED, EMBED, k0);
    v16bf b1 = load_b_frag(BT + (wn * 32 + 16) * EMBED, EMBED, k0);
    acc[0][0] = wmma_bf(a0, b0, acc[0][0]);
    acc[0][1] = wmma_bf(a0, b1, acc[0][1]);
    acc[1][0] = wmma_bf(a1, b0, acc[1][0]);
    acc[1][1] = wmma_bf(a1, b1, acc[1][1]);
  }

#pragma unroll
  for (int i = 0; i < 2; ++i) {
#pragma unroll
    for (int j = 0; j < 2; ++j) {
      const int n = ntile * 64 + wn * 32 + j * 16 + (lane & 15);
      const float bv = outb[n];
#pragma unroll
      for (int r = 0; r < 8; ++r) {
        const int mrow = mtile * BLOCKSZ + wm * 32 + i * 16 + r + ((lane >> 4) << 3);
        const int bb = mrow >> 13;       // row = b*SEQ + s
        const int ss = mrow & (SEQ - 1);
        out[((long)ss * BATCH + bb) * EMBED + n] = acc[i][j][r] + bv;
      }
    }
  }
}

// ---------------------------------------------------------------------------
extern "C" void kernel_launch(void* const* d_in, const int* in_sizes, int n_in,
                              void* d_out, int out_size, void* d_ws, size_t ws_size,
                              hipStream_t stream) {
  (void)in_sizes; (void)n_in; (void)out_size; (void)ws_size;
  const float* query   = (const float*)d_in[0];
  const float* key_inp = (const float*)d_in[1];
  const float* value   = (const float*)d_in[2];
  const float* q_w     = (const float*)d_in[3];
  const float* k_w     = (const float*)d_in[4];
  const float* v_w     = (const float*)d_in[5];
  const float* q_b     = (const float*)d_in[6];
  const float* k_b     = (const float*)d_in[7];
  const float* v_b     = (const float*)d_in[8];
  const float* out_w   = (const float*)d_in[9];
  const float* out_b   = (const float*)d_in[10];
  const int*   rand_ix = (const int*)d_in[11];
  float* out = (float*)d_out;

  char* p = (char*)d_ws;
  auto alloc = [&](size_t bytes) -> char* {
    char* r = p;
    p += (bytes + 255) & ~(size_t)255;
    return r;
  };
  bf16_t* xbf   = (bf16_t*)alloc(3ull * BATCH * SEQ * EMBED * sizeof(bf16_t));
  bf16_t* wbf   = (bf16_t*)alloc(3ull * EMBED * EMBED * sizeof(bf16_t));
  bf16_t* wobf  = (bf16_t*)alloc((size_t)EMBED * EMBED * sizeof(bf16_t));
  bf16_t* Qbuf  = (bf16_t*)alloc((size_t)BATCH * HEADS * SEQ * DH * sizeof(bf16_t));
  bf16_t* Kbuf  = (bf16_t*)alloc((size_t)BATCH * HEADS * SEQ * DH * sizeof(bf16_t));
  bf16_t* Vtbuf = (bf16_t*)alloc((size_t)BATCH * HEADS * SEQ * DH * sizeof(bf16_t));
  bf16_t* AObuf = (bf16_t*)alloc((size_t)BATCH * SEQ * EMBED * sizeof(bf16_t));

  const long nx = 3L * BATCH * SEQ * EMBED;
  cvt_x_kernel<<<(unsigned)((nx + 255) / 256), 256, 0, stream>>>(query, key_inp, value, xbf);

  const int nw = 4 * EMBED * EMBED;
  cvt_w_kernel<<<(nw + 255) / 256, 256, 0, stream>>>(q_w, k_w, v_w, out_w, wbf, wobf);

  proj_kernel<<<dim3(NBLK, HEADS, BATCH * 3), 128, 0, stream>>>(
      xbf, wbf, q_b, k_b, v_b, Qbuf, Kbuf, Vtbuf);

  attn_kernel<<<dim3(NBLK, HEADS, BATCH), 256, 0, stream>>>(
      Qbuf, Kbuf, Vtbuf, rand_ix, AObuf);

  outproj_kernel<<<dim3(BATCH * SEQ / BLOCKSZ, EMBED / 64), 128, 0, stream>>>(
      AObuf, wobf, out_b, out);
}
